// FinalCALMDecoderLayer_35983236006607
// MI455X (gfx1250) — compile-verified, matrix operand order checked by reading, not run on hardware
//
#include <hip/hip_runtime.h>
#include <math.h>

// ---------------------------------------------------------------------------
// CDNA5 (gfx1250, wave32) implementation.
// Pipeline:
//   K0 build_w2pp : W2''(3072x32) f16  = [W2 rearranged | filt | zeros]
//   K1 select     : per-query knn(103 of 2048), softmax weights, RFF-MLP h,
//                   emits hk'(48x128) f16 (zero-padded K) + gather indices
//   K2 xl_gemm    : xl16 = f16(x @ W_lin^T + b_lin)        [WMMA f16]
//   K3 yq_gemm    : per-q Y = hk' @ gathered(xl16)         [WMMA f16,
//                   A-panel staged via global_load_async_to_lds_b128]
//   K4 out_gemm   : out = Y @ W2'' + bias (f32 out)        [WMMA f16,
//                   B-panel staged through one LDS buffer in 64KB async chunks]
// Workspace: ~34 MB (W2'' 192KB, hk' 6MB, ind 256KB, xl16 2MB, Y 24MB).
// ---------------------------------------------------------------------------

typedef __attribute__((ext_vector_type(16))) _Float16 v16h;
typedef __attribute__((ext_vector_type(8)))  float    v8f;

#define QN   512
#define VN   2048
#define KSEL 103
#define KPAD 128
#define JROW 48     // 32 h-channels + 1 kd-channel + 15 zero pad (3 M-tiles)
#define BT   8      // B*T
#define YK   (JROW*64)  // 3072

__device__ __forceinline__ v8f zero8() {
  v8f z = {0.f,0.f,0.f,0.f,0.f,0.f,0.f,0.f};
  return z;
}

// Async copy of 16 bytes global->LDS (ASYNCcnt-tracked, no VGPR data path).
// ldsoff = logical LDS byte offset (low 32 bits of generic pointer to LDS).
__device__ __forceinline__ void async_g2l_b128(unsigned ldsoff, const void* g) {
  asm volatile("global_load_async_to_lds_b128 %0, %1, off"
               :: "v"(ldsoff), "v"((unsigned long long)(size_t)g)
               : "memory");
}
__device__ __forceinline__ void wait_async0() {
  asm volatile("s_wait_asynccnt 0x0" ::: "memory");
}

// A fragment (16x32 f16) from row-major f16 src, leading dim ld.
// ISA 7.12.2: lanes<16 -> row M=lane, K = kbase+{0..7,16..23};
//             lanes>=16 -> K = kbase+{8..15,24..31}.
__device__ __forceinline__ v16h loadA_h(const _Float16* src, int ld, int lane,
                                        int rbase, int kbase) {
  const int r  = rbase + (lane & 15);
  const int kb = kbase + ((lane & 16) ? 8 : 0);
  const _Float16* p = src + (size_t)r * ld + kb;
  v16h a;
#pragma unroll
  for (int e = 0; e < 8; ++e) a[e] = p[e];
#pragma unroll
  for (int e = 0; e < 8; ++e) a[8 + e] = p[16 + e];
  return a;
}

// ---------------------------------------------------------------------------
// K0: W2''[(j*64+c)*32+d] = j<32 ? W2[(c*32+d)*32+j] : (j==32 ? filt[c*32+d] : 0)
__global__ void build_w2pp(const float* __restrict__ W2,
                           const float* __restrict__ filt,
                           _Float16* __restrict__ W2pp) {
  int idx = blockIdx.x * blockDim.x + threadIdx.x;   // [0, 3072*32)
  int row = idx >> 5;          // j*64 + c
  int d   = idx & 31;
  int j   = row >> 6;
  int c   = row & 63;
  float v;
  if (j < 32)       v = W2[(c * 32 + d) * 32 + j];
  else if (j == 32) v = filt[c * 32 + d];
  else              v = 0.f;
  W2pp[idx] = (_Float16)v;
}

// ---------------------------------------------------------------------------
// K1: per-query selection + weights + h-MLP. Block = 128 thr, grid = 512.
__global__ void select_build(const float* __restrict__ pos,
                             const float* __restrict__ qpos,
                             const float* __restrict__ Bmat,
                             const float* __restrict__ W1,
                             const float* __restrict__ b1,
                             _Float16* __restrict__ hk16,
                             int* __restrict__ indw) {
  __shared__ float s_e[VN];
  __shared__ int   s_v[KPAD];
  __shared__ float s_se[KPAD];
  __shared__ float s_w[KPAD];
  __shared__ float s_stat[3];
  const int q = blockIdx.x, tid = threadIdx.x;
  const float qx = qpos[q * 2 + 0], qy = qpos[q * 2 + 1];

  for (int v = tid; v < VN; v += 128) {
    float dx = qx - pos[v * 2 + 0] + 0.5f; dx -= floorf(dx); dx -= 0.5f;
    float dy = qy - pos[v * 2 + 1] + 0.5f; dy -= floorf(dy); dy -= 0.5f;
    s_e[v] = dx * dx + dy * dy;
  }
  if (tid >= KSEL && tid < KPAD) { s_v[tid] = 0; s_se[tid] = 0.f; }
  __syncthreads();

  // exact rank-by-counting: rank(v) = #{j : e_j < e_v or (e_j==e_v and j<v)}
  for (int v = tid; v < VN; v += 128) {
    const float ev = s_e[v];
    int rank = 0;
    for (int j = 0; j < VN; ++j) {
      const float ej = s_e[j];
      rank += (ej < ev) || (ej == ev && j < v);
    }
    if (rank < KSEL) { s_v[rank] = v; s_se[rank] = ev; }
  }
  __syncthreads();

  if (tid == 0) {
    float mn = 3.4e38f, mx = -3.4e38f;
    for (int k = 0; k < KSEL; ++k) { mn = fminf(mn, s_se[k]); mx = fmaxf(mx, s_se[k]); }
    s_stat[0] = mn; s_stat[1] = mx;
  }
  __syncthreads();
  if (tid < KSEL) {
    float en = (s_se[tid] - s_stat[0]) / ((s_stat[1] - s_stat[0]) + 1e-8f);
    s_w[tid] = expf(-en);
  } else if (tid < KPAD) s_w[tid] = 0.f;
  __syncthreads();
  if (tid == 0) {
    float s = 0.f;
    for (int k = 0; k < KSEL; ++k) s += s_w[k];
    s_stat[2] = s;
  }
  __syncthreads();

  if (tid < KPAD) {
    _Float16* hq = hk16 + (size_t)q * JROW * KPAD;
    if (tid < KSEL) {
      const int v = s_v[tid];
      const float kd = s_w[tid] / s_stat[2];
      float dx = qx - pos[v * 2 + 0] + 0.5f; dx -= floorf(dx); dx -= 0.5f;
      float dy = qy - pos[v * 2 + 1] + 0.5f; dy -= floorf(dy); dy -= 0.5f;
      float kf[32];
#pragma unroll
      for (int r = 0; r < 16; ++r) {
        float pr = 6.2831853071795864f * (dx * Bmat[r] + dy * Bmat[16 + r]);
        kf[r] = sinf(pr); kf[16 + r] = cosf(pr);
      }
      for (int j = 0; j < 32; ++j) {
        float acc = b1[j];
#pragma unroll
        for (int r = 0; r < 32; ++r) acc += kf[r] * W1[j * 32 + r];
        float h = 0.5f * acc * (1.f + erff(acc * 0.70710678118f));  // exact gelu
        hq[j * KPAD + tid] = (_Float16)(h * kd);
      }
      hq[32 * KPAD + tid] = (_Float16)kd;          // carries the filt term
      for (int j = 33; j < JROW; ++j) hq[j * KPAD + tid] = (_Float16)0.f;
      indw[q * KPAD + tid] = v;
    } else {
      for (int j = 0; j < JROW; ++j) hq[j * KPAD + tid] = (_Float16)0.f;
      indw[q * KPAD + tid] = 0;   // hk' column is zero -> contributes nothing
    }
  }
}

// ---------------------------------------------------------------------------
// K2: xl16 = f16(x @ W_lin^T + b_lin). 16384x64x64. Block=128 (4 waves),
// each wave owns a 16x64 output tile. Grid = 256.
__global__ void xl_gemm(const float* __restrict__ x,
                        const float* __restrict__ W_lin,
                        const float* __restrict__ b_lin,
                        _Float16* __restrict__ xl16) {
  const int lane = threadIdx.x & 31, wave = threadIdx.x >> 5;
  const int rowBase = (blockIdx.x * 4 + wave) * 16;
  const float* xrow = x + (size_t)rowBase * 64;
  v8f acc[4];
#pragma unroll
  for (int nt = 0; nt < 4; ++nt) acc[nt] = zero8();

#pragma unroll
  for (int kt = 0; kt < 2; ++kt) {
    const int r  = lane & 15;
    const int kb = kt * 32 + ((lane & 16) ? 8 : 0);
    v16h a;
    const float* pa = xrow + r * 64 + kb;
#pragma unroll
    for (int e = 0; e < 8; ++e) a[e] = (_Float16)pa[e];
#pragma unroll
    for (int e = 0; e < 8; ++e) a[8 + e] = (_Float16)pa[16 + e];
#pragma unroll
    for (int nt = 0; nt < 4; ++nt) {
      const int n = nt * 16 + (lane & 15);
      v16h b;                                  // B[k][n] = W_lin[n][k]
      const float* pb = W_lin + n * 64 + kb;
#pragma unroll
      for (int e = 0; e < 8; ++e) b[e] = (_Float16)pb[e];
#pragma unroll
      for (int e = 0; e < 8; ++e) b[8 + e] = (_Float16)pb[16 + e];
      acc[nt] = __builtin_amdgcn_wmma_f32_16x16x32_f16(
          false, a, false, b, (short)0, acc[nt], false, false);
    }
  }
  const int mo = (lane & 16) ? 8 : 0, n0 = lane & 15;
#pragma unroll
  for (int nt = 0; nt < 4; ++nt) {
    const int col = nt * 16 + n0;
    const float bl = b_lin[col];
#pragma unroll
    for (int i = 0; i < 8; ++i)
      xl16[(size_t)(rowBase + mo + i) * 64 + col] = (_Float16)(acc[nt][i] + bl);
  }
}

// ---------------------------------------------------------------------------
// K3: per-query Y(48 x 512) = hk'(48x128) @ gathered xl (128x512),
// N index = bt*64 + c; wave w handles bt=w. Block=256 (8 waves), grid=512.
// hk' A-panel (12 KB) staged once per block via async-to-LDS, shared by all
// 8 waves (was: 8x redundant global reloads).
__global__ void yq_gemm(const _Float16* __restrict__ xl16,
                        const _Float16* __restrict__ hk16,
                        const int* __restrict__ indw,
                        _Float16* __restrict__ Y) {
  __shared__ __align__(16) _Float16 sHQ[JROW * KPAD];   // 12 KB
  __shared__ int s_ind[KPAD];
  const int q = blockIdx.x, tid = threadIdx.x;
  const int lane = tid & 31, wave = tid >> 5;   // wave == bt
  if (tid < KPAD) s_ind[tid] = indw[q * KPAD + tid];

  const _Float16* hq = hk16 + (size_t)q * JROW * KPAD;
#pragma unroll
  for (int it = 0; it < (JROW * KPAD) / (256 * 8); ++it) {   // 3 iters
    const int elt = (it * 256 + tid) * 8;                    // 8 f16 = 16 B
    async_g2l_b128((unsigned)(size_t)(void*)&sHQ[elt], hq + elt);
  }
  wait_async0();
  __syncthreads();

  const size_t btbase = (size_t)wave * VN;
  v8f acc[3][4];
#pragma unroll
  for (int mt = 0; mt < 3; ++mt)
#pragma unroll
    for (int nt = 0; nt < 4; ++nt) acc[mt][nt] = zero8();

#pragma unroll
  for (int kt = 0; kt < 4; ++kt) {
    v16h a[3];
#pragma unroll
    for (int mt = 0; mt < 3; ++mt) a[mt] = loadA_h(sHQ, KPAD, lane, mt * 16, kt * 32);
    const int kb = kt * 32 + ((lane & 16) ? 8 : 0);
#pragma unroll
    for (int nt = 0; nt < 4; ++nt) {
      const int c = nt * 16 + (lane & 15);
      v16h b;                                  // gathered B[k][n] = xl16[bt, ind[k], c]
#pragma unroll
      for (int e = 0; e < 8; ++e) {
        b[e]     = xl16[(btbase + (size_t)s_ind[kb + e]) * 64 + c];
        b[8 + e] = xl16[(btbase + (size_t)s_ind[kb + 16 + e]) * 64 + c];
      }
#pragma unroll
      for (int mt = 0; mt < 3; ++mt)
        acc[mt][nt] = __builtin_amdgcn_wmma_f32_16x16x32_f16(
            false, a[mt], false, b, (short)0, acc[mt][nt], false, false);
    }
  }
  const size_t yrow = (size_t)(wave * QN + q) * YK;
  const int mo = (lane & 16) ? 8 : 0, n0 = lane & 15;
#pragma unroll
  for (int mt = 0; mt < 3; ++mt)
#pragma unroll
    for (int nt = 0; nt < 4; ++nt)
#pragma unroll
      for (int i = 0; i < 8; ++i) {
        const int j = mt * 16 + mo + i;
        Y[yrow + j * 64 + nt * 16 + n0] = (_Float16)acc[mt][nt][i];
      }
}

// ---------------------------------------------------------------------------
// K4: out(4096x32) = Y(4096x3072) @ W2''(3072x32) + bias. Block=128, grid=64.
// W2'' streamed through LDS in three 64 KB async chunks (1024 k-rows each);
// each chunk is reused by all 4 waves for 32 k-steps.
__global__ void out_gemm(const _Float16* __restrict__ Y,
                         const _Float16* __restrict__ W2pp,
                         const float* __restrict__ bias,
                         float* __restrict__ out) {
  __shared__ __align__(16) _Float16 sW2[1024 * 32];   // 64 KB chunk
  const int lane = threadIdx.x & 31, wave = threadIdx.x >> 5;
  const int rowBase = (blockIdx.x * 4 + wave) * 16;
  const _Float16* yrow = Y + (size_t)rowBase * YK;
  v8f acc[2] = {zero8(), zero8()};

  for (int ck = 0; ck < 3; ++ck) {
    // stage 32768 f16 (64 KB): 128 threads x 32 iters x 16 B, ASYNCcnt<=32
#pragma unroll 1
    for (int it = 0; it < 32; ++it) {
      const int elt = (it * 128 + threadIdx.x) * 8;
      async_g2l_b128((unsigned)(size_t)(void*)&sW2[elt],
                     W2pp + (size_t)ck * 32768 + elt);
    }
    wait_async0();
    __syncthreads();

#pragma unroll 1
    for (int ksl = 0; ksl < 32; ++ksl) {
      const int ks = ck * 32 + ksl;
      v16h a = loadA_h(yrow, YK, lane, 0, ks * 32);
      const int kb = ksl * 32 + ((lane & 16) ? 8 : 0);   // k-row local to chunk
#pragma unroll
      for (int nt = 0; nt < 2; ++nt) {
        const int n = nt * 16 + (lane & 15);
        v16h b;
#pragma unroll
        for (int e = 0; e < 8; ++e) {
          b[e]     = sW2[(kb + e) * 32 + n];
          b[8 + e] = sW2[(kb + 16 + e) * 32 + n];
        }
        acc[nt] = __builtin_amdgcn_wmma_f32_16x16x32_f16(
            false, a, false, b, (short)0, acc[nt], false, false);
      }
    }
    __syncthreads();   // before next chunk overwrites sW2
  }

  const int mo = (lane & 16) ? 8 : 0, n0 = lane & 15;
#pragma unroll
  for (int nt = 0; nt < 2; ++nt) {
    const int col = nt * 16 + n0;
    const float bv = bias[col];
#pragma unroll
    for (int i = 0; i < 8; ++i)
      out[(size_t)(rowBase + mo + i) * 32 + col] = acc[nt][i] + bv;
  }
}

// ---------------------------------------------------------------------------
extern "C" void kernel_launch(void* const* d_in, const int* in_sizes, int n_in,
                              void* d_out, int out_size, void* d_ws, size_t ws_size,
                              hipStream_t stream) {
  (void)in_sizes; (void)n_in; (void)out_size; (void)ws_size;
  const float* x     = (const float*)d_in[0];
  const float* pos   = (const float*)d_in[1];
  const float* qpos  = (const float*)d_in[2];
  const float* W_lin = (const float*)d_in[3];
  const float* b_lin = (const float*)d_in[4];
  const float* Bmat  = (const float*)d_in[5];
  const float* W1    = (const float*)d_in[6];
  const float* b1    = (const float*)d_in[7];
  const float* W2    = (const float*)d_in[8];
  const float* filt  = (const float*)d_in[9];
  const float* bias  = (const float*)d_in[10];
  float* out = (float*)d_out;

  char* ws = (char*)d_ws;
  size_t off = 0;
  auto carve = [&](size_t bytes) -> void* {
    void* p = ws + off;
    off += (bytes + 255) & ~(size_t)255;
    return p;
  };
  _Float16* W2pp = (_Float16*)carve((size_t)YK * 32 * sizeof(_Float16));          // 192 KB
  _Float16* hk16 = (_Float16*)carve((size_t)QN * JROW * KPAD * sizeof(_Float16)); // 6 MB
  int*      indw = (int*)carve((size_t)QN * KPAD * sizeof(int));                  // 256 KB
  _Float16* xl16 = (_Float16*)carve((size_t)BT * VN * 64 * sizeof(_Float16));     // 2 MB
  _Float16* Yb   = (_Float16*)carve((size_t)BT * QN * YK * sizeof(_Float16));     // 24 MB

  build_w2pp  <<<(YK * 32) / 256, 256, 0, stream>>>(W2, filt, W2pp);
  select_build<<<QN, 128, 0, stream>>>(pos, qpos, Bmat, W1, b1, hk16, indw);
  xl_gemm     <<<(BT * VN) / 64, 128, 0, stream>>>(x, W_lin, b_lin, xl16);
  yq_gemm     <<<QN, 256, 0, stream>>>(xl16, hk16, indw, Yb);
  out_gemm    <<<(BT * QN) / 64, 128, 0, stream>>>(Yb, W2pp, bias, out);
}
